// ExplicitConditioningDecoder_7636451852790
// MI455X (gfx1250) — matrix operand order repaired and checked by reading, not compiled, not run on hardware
//
#include <hip/hip_runtime.h>

// Problem constants (from reference)
#define V_ 32000
#define E_ 512
#define H_ 1024
#define T_ 64
#define B_ 32
#define S_ 32
#define XW_ (E_ + H_)        // 1536  (GRU input width)
#define GW_ (3 * H_)         // 3072  (gate width)
#define FW_ (E_ + 2 * H_)    // 2560  (output feature width)
#define MROWS_ (T_ * B_)     // 2048

typedef __attribute__((ext_vector_type(16))) __bf16        v16bf;
typedef __attribute__((ext_vector_type(8)))  float         v8f;
typedef __attribute__((ext_vector_type(8)))  unsigned int  u32x8;
typedef __attribute__((ext_vector_type(4)))  unsigned int  u32x4;

__device__ __forceinline__ unsigned short f32_to_bf16_rne(float f) {
    union { float f; unsigned int u; } c; c.f = f;
    unsigned int u = c.u;
    if ((u & 0x7fffffffu) > 0x7f800000u)          // NaN: quiet it
        return (unsigned short)((u >> 16) | 0x40);
    u += 0x7fffu + ((u >> 16) & 1u);              // round-to-nearest-even
    return (unsigned short)(u >> 16);
}

__device__ __forceinline__ v16bf load_a_frag(const unsigned short* p, int k) {
    // A 16x32 bf16 frag (ISA 7.12.2): lane row m=l&15; VGPR0-3 hold
    // K in [ (l>>4)*8, +8 ), VGPR4-7 hold K in [ 16+(l>>4)*8, +8 ).
    u32x4 lo = *(const u32x4*)(p + k);
    u32x4 hi = *(const u32x4*)(p + k + 16);
    u32x8 a8;
    a8[0] = lo[0]; a8[1] = lo[1]; a8[2] = lo[2]; a8[3] = lo[3];
    a8[4] = hi[0]; a8[5] = hi[1]; a8[6] = hi[2]; a8[7] = hi[3];
    return __builtin_bit_cast(v16bf, a8);
}

// ---------------------------------------------------------------- converts
__global__ void convert_f32_bf16(const float* __restrict__ in,
                                 unsigned short* __restrict__ out, size_t n) {
    size_t i = blockIdx.x * (size_t)blockDim.x + threadIdx.x;
    size_t stride = (size_t)gridDim.x * blockDim.x;
    for (; i < n; i += stride) out[i] = f32_to_bf16_rne(in[i]);
}

// ------------------------------------------- build x = [emb|ctx] and feats
__global__ void build_x_feats(const int* __restrict__ ids,      // T*B
                              const float* __restrict__ emb,    // V x E
                              const float* __restrict__ enc,    // S x B x H
                              unsigned short* __restrict__ x,     // (T*B) x XW
                              unsigned short* __restrict__ feats) // (T*B) x FW
{
    size_t total = (size_t)MROWS_ * XW_;
    size_t i = blockIdx.x * (size_t)blockDim.x + threadIdx.x;
    size_t stride = (size_t)gridDim.x * blockDim.x;
    for (; i < total; i += stride) {
        int c = (int)(i % XW_);
        size_t row = i / XW_;
        int b = (int)(row % B_);
        int t = (int)(row / B_);
        float v;
        int fcol;
        if (c < E_) {
            int id = (t == 0) ? ids[b] : ids[(size_t)(t - 1) * B_ + b];
            v = emb[(size_t)id * E_ + c];
            fcol = c;                                   // emb slot [0,512)
        } else {
            v = enc[(size_t)(S_ - 1) * B_ * H_ + (size_t)b * H_ + (c - E_)];
            fcol = c + H_;                              // ctx slot [1536,2560)
        }
        unsigned short bv = f32_to_bf16_rne(v);
        x[row * XW_ + c] = bv;
        feats[row * (size_t)FW_ + fcol] = bv;
    }
}

// ------------------------------------------------------- WMMA bf16 GEMM
// C[M][N] = A[M][K] (bf16 row-major) @ W[N][K]^T (bf16 row-major) + bias
// Register-blocked: each wave computes a 32x64 tile of C (2 Mtiles x 4
// Ntiles, 8 v8f accumulators). Per K-step of 32: 12 b128 loads -> 8 WMMAs.
// All launch dims divide exactly (no tails), so EXEC is all-ones for WMMA.
#define WPB 4                                   // waves per block (along N)
__global__ __launch_bounds__(32 * WPB)
void wmma_gemm_bf16_nt(const unsigned short* __restrict__ A,
                       const unsigned short* __restrict__ W,
                       const float* __restrict__ bias,
                       float* __restrict__ C,
                       int M, int N, int K) {
    const int lane  = threadIdx.x;              // 0..31
    const int nbase = (blockIdx.x * WPB + threadIdx.y) * 64;
    const int mbase = blockIdx.y * 32;
    if (nbase >= N || mbase >= M) return;       // uniform per wave

    const int half = lane >> 4;                 // 0 or 1
    const int lm   = lane & 15;

    // Accumulators, initialized with the (broadcast-per-column) bias.
    v8f acc[2][4];
#pragma unroll
    for (int ni = 0; ni < 4; ++ni) {
        float bv = bias ? bias[nbase + ni * 16 + lm] : 0.0f;
#pragma unroll
        for (int mi = 0; mi < 2; ++mi)
#pragma unroll
            for (int j = 0; j < 8; ++j) acc[mi][ni][j] = bv;
    }

    // A frag base: lane row m = lane&15 (+16 for second Mtile)
    const unsigned short* arow0 = A + (size_t)(mbase + lm) * K + half * 8;
    const unsigned short* arow1 = arow0 + (size_t)16 * K;
    // B frag base: lane col n; K chunks [half*8, +8) and [16+half*8, +8) in W row n
    const unsigned short* brow[4];
#pragma unroll
    for (int ni = 0; ni < 4; ++ni)
        brow[ni] = W + (size_t)(nbase + ni * 16 + lm) * K + half * 8;

    for (int k = 0; k < K; k += 32) {
        v16bf a0 = load_a_frag(arow0, k);
        v16bf a1 = load_a_frag(arow1, k);
#pragma unroll
        for (int ni = 0; ni < 4; ++ni) {
            // Prefetch the HBM-resident W stream ~2KB (32 K-steps) ahead.
            __builtin_prefetch(brow[ni] + k + 1024, 0, 1);
            v16bf bf = load_a_frag(brow[ni], k);
            acc[0][ni] = __builtin_amdgcn_wmma_f32_16x16x32_bf16(
                false, a0, false, bf, (short)0, acc[0][ni], false, false);
            acc[1][ni] = __builtin_amdgcn_wmma_f32_16x16x32_bf16(
                false, a1, false, bf, (short)0, acc[1][ni], false, false);
        }
    }

    // C/D layout: VGPR j -> row m = 8*half + j, col n = lane&15
#pragma unroll
    for (int mi = 0; mi < 2; ++mi) {
        float* crow = C + (size_t)(mbase + mi * 16 + half * 8) * N + nbase + lm;
#pragma unroll
        for (int ni = 0; ni < 4; ++ni)
#pragma unroll
            for (int j = 0; j < 8; ++j)
                crow[(size_t)j * N + ni * 16] = acc[mi][ni][j];
    }
}

// ----------------------------------------------------------- GRU pointwise
__global__ void gru_gate(const float* __restrict__ x_proj,   // (T*B) x 3H
                         const float* __restrict__ gh,       // B x 3H (has b_hh)
                         float* __restrict__ h,              // B x H f32
                         unsigned short* __restrict__ h_bf,  // B x H bf16
                         unsigned short* __restrict__ feats, // (T*B) x FW
                         int t) {
    int i = blockIdx.x * blockDim.x + threadIdx.x;
    if (i >= B_ * H_) return;
    int b = i / H_, j = i % H_;
    const float* xp = x_proj + ((size_t)t * B_ + b) * GW_;
    const float* gp = gh + (size_t)b * GW_;
    float r = 1.0f / (1.0f + __expf(-(xp[j] + gp[j])));
    float z = 1.0f / (1.0f + __expf(-(xp[H_ + j] + gp[H_ + j])));
    float nn = tanhf(xp[2 * H_ + j] + r * gp[2 * H_ + j]);
    float hnew = (1.0f - z) * nn + z * h[i];
    h[i] = hnew;
    unsigned short hb = f32_to_bf16_rne(hnew);
    h_bf[i] = hb;
    feats[((size_t)t * B_ + b) * FW_ + E_ + j] = hb;   // hs slot [512,1536)
}

__global__ void zero_h(float* __restrict__ h, unsigned short* __restrict__ hb) {
    int i = blockIdx.x * blockDim.x + threadIdx.x;
    if (i < B_ * H_) { h[i] = 0.0f; hb[i] = 0; }
}

__global__ void copy_hidden(const float* __restrict__ h, float* __restrict__ out) {
    int i = blockIdx.x * blockDim.x + threadIdx.x;
    if (i < B_ * H_) out[i] = h[i];
}

// ----------------------------------------------------------------- driver
extern "C" void kernel_launch(void* const* d_in, const int* in_sizes, int n_in,
                              void* d_out, int out_size, void* d_ws, size_t ws_size,
                              hipStream_t stream) {
    const int*   inputs = (const int*)d_in[0];
    const float* enc    = (const float*)d_in[1];
    // d_in[2] = encoder_mask (unused by reference)
    const float* emb    = (const float*)d_in[3];
    const float* W_ih   = (const float*)d_in[4];
    const float* W_hh   = (const float*)d_in[5];
    const float* b_ih   = (const float*)d_in[6];
    const float* b_hh   = (const float*)d_in[7];
    const float* W_out  = (const float*)d_in[8];
    const float* b_out  = (const float*)d_in[9];

    float* logits     = (float*)d_out;                          // T*B*V
    float* hidden_out = logits + (size_t)T_ * B_ * V_;          // B*H

    char* ws = (char*)d_ws;
    size_t off = 0;
    auto salloc = [&](size_t bytes) -> void* {
        void* p = ws + off;
        off += (bytes + 255) & ~(size_t)255;
        return p;
    };
    unsigned short* W_ih_bf  = (unsigned short*)salloc((size_t)GW_ * XW_ * 2);
    unsigned short* W_hh_bf  = (unsigned short*)salloc((size_t)GW_ * H_ * 2);
    unsigned short* W_out_bf = (unsigned short*)salloc((size_t)V_ * FW_ * 2);
    unsigned short* x_bf     = (unsigned short*)salloc((size_t)MROWS_ * XW_ * 2);
    unsigned short* feats_bf = (unsigned short*)salloc((size_t)MROWS_ * FW_ * 2);
    float*          x_proj   = (float*)salloc((size_t)MROWS_ * GW_ * 4);
    float*          gh       = (float*)salloc((size_t)B_ * GW_ * 4);
    float*          h        = (float*)salloc((size_t)B_ * H_ * 4);
    unsigned short* h_bf     = (unsigned short*)salloc((size_t)B_ * H_ * 2);
    (void)ws_size; (void)in_sizes; (void)n_in; (void)out_size;

    // 1) weights -> bf16
    convert_f32_bf16<<<2048, 256, 0, stream>>>(W_ih,  W_ih_bf,  (size_t)GW_ * XW_);
    convert_f32_bf16<<<2048, 256, 0, stream>>>(W_hh,  W_hh_bf,  (size_t)GW_ * H_);
    convert_f32_bf16<<<4096, 256, 0, stream>>>(W_out, W_out_bf, (size_t)V_ * FW_);

    // 2) gather embeddings + broadcast context -> x (bf16) and feats slots
    build_x_feats<<<4096, 256, 0, stream>>>(inputs, emb, enc, x_bf, feats_bf);

    // 3) x_proj = x @ W_ih^T + b_ih      (2048 x 1536) @ (1536 x 3072)
    {
        dim3 blk(32, WPB), grd(GW_ / (64 * WPB), MROWS_ / 32);   // (12, 64)
        wmma_gemm_bf16_nt<<<grd, blk, 0, stream>>>(x_bf, W_ih_bf, b_ih, x_proj,
                                                   MROWS_, GW_, XW_);
    }

    // 4) GRU scan (sequential chain of WMMA step-GEMMs + fused gates)
    zero_h<<<(B_ * H_ + 255) / 256, 256, 0, stream>>>(h, h_bf);
    for (int t = 0; t < T_; ++t) {
        dim3 blk(32, WPB), grd(GW_ / (64 * WPB), 1);             // (12, 1)
        wmma_gemm_bf16_nt<<<grd, blk, 0, stream>>>(h_bf, W_hh_bf, b_hh, gh,
                                                   B_, GW_, H_);
        gru_gate<<<(B_ * H_ + 255) / 256, 256, 0, stream>>>(x_proj, gh, h, h_bf,
                                                            feats_bf, t);
    }

    // 5) logits = feats @ W_out^T + b_out   (2048 x 2560) @ (2560 x 32000)
    {
        dim3 blk(32, WPB), grd(V_ / (64 * WPB), MROWS_ / 32);    // (125, 64)
        wmma_gemm_bf16_nt<<<grd, blk, 0, stream>>>(feats_bf, W_out_bf, b_out,
                                                   logits, MROWS_, V_, FW_);
    }

    // 6) final hidden state
    copy_hidden<<<(B_ * H_ + 255) / 256, 256, 0, stream>>>(h, hidden_out);
}
